// L1AttnSparseBidi_82738249990760
// MI455X (gfx1250) — compile-verified, matrix operand order checked by reading, not run on hardware
//
#include <hip/hip_runtime.h>
#include <math.h>

// ---------------------------------------------------------------------------
// L1AttnSparseBidi for MI455X (gfx1250, wave32)
//   bs=2, heads=8, width=32 are structural constants of the reference.
//   n_tok and E (edge count) are derived from in_sizes at launch.
//
// Pipeline (3 kernels, all on `stream`):
//   0) zero vo (d_out) and denom[bs][n_tok][8]
//   1) ww_kernel: per (b,edge): L1 distances for all 8 heads via one
//      v_wmma_f32_16x16x32_f16 row-sum reduction (A = |q-k| in f16,
//      B = ones, f32 accumulate). Rows staged to LDS with async
//      global->LDS b128 loads (ASYNCcnt). Writes W[b][e][h] = exp(ww),
//      atomically accumulates denom[b][dst][h].  Max-free softmax is
//      numerically safe: all logits <= 0 so exp in (0,1].
//   2) scatter_kernel: per (b,edge): wgt = W/denom(dst); hardware f32
//      atomics: vo[dst] += wgt*vf[src], vo[src] += wgt*vb[dst].
//
// NOTE: an empty asm "+v" barrier is applied to each q-k difference before
// fabsf: the gfx1250 backend crashes in ISel when SLP packs the differences
// into v2f32 and the fabs combine emits an ill-typed scalar-int mask node.
// The barrier keeps the abs scalar and sidesteps the bug at zero cost.
// ---------------------------------------------------------------------------

typedef __attribute__((ext_vector_type(16))) _Float16 v16h;
typedef __attribute__((ext_vector_type(8)))  float    v8f;
typedef __attribute__((ext_vector_type(4)))  float    f32x4;
// Exact parameter type of __builtin_amdgcn_global_load_async_to_lds_b128
// per hipcc diagnostic: 'int __attribute__((vector_size(4*sizeof(int)))) *'
typedef int v4i_vs __attribute__((vector_size(16)));

#define HEADS 8
#define WIDTH 32
#define BS    2
#define ROWF  (HEADS * WIDTH)   // 256 floats per (b,tok) row
#define ROWP  264               // padded LDS row stride in floats
#define WPB   8                 // waves per block (256 threads, wave32)
#define JPW   2                 // (b,edge) jobs per wave -> 16 WMMA rows

#if defined(__has_builtin)
#if __has_builtin(__builtin_amdgcn_global_load_async_to_lds_b128)
#define HAVE_ASYNC_LDS 1
#endif
#endif

__global__ void l1sb_zero_kernel(float* __restrict__ p, long n) {
  long i = (long)blockIdx.x * blockDim.x + threadIdx.x;
  long stride = (long)gridDim.x * blockDim.x;
  for (; i < n; i += stride) p[i] = 0.0f;
}

__global__ void __launch_bounds__(256)
l1sb_ww_kernel(const float* __restrict__ q, const float* __restrict__ kk,
               const int* __restrict__ coo, float* __restrict__ W,
               float* __restrict__ denom, int nTok, int E) {
  __shared__ float sm[WPB * 4 * ROWP];   // 4 rows x 1KB per wave (+pad)
  const int wave = threadIdx.x >> 5;
  const int lane = threadIdx.x & 31;
  const long nJobs = (long)BS * E;
  const long base  = ((long)blockIdx.x * WPB + wave) * JPW;

  bool valid[2];
  int  bidx[2], eidx[2], dtok[2], stok[2];
#pragma unroll
  for (int t = 0; t < 2; ++t) {
    long j = base + t;
    valid[t] = (j < nJobs);
    long c = valid[t] ? j : (nJobs - 1);   // clamp: keep EXEC all-1s for WMMA
    int b = (int)(c / E);
    int e = (int)(c - (long)b * E);
    bidx[t] = b;
    eidx[t] = e;
    dtok[t] = coo[3 * e + 0];
    stok[t] = coo[3 * e + 1];
  }

  float* wavebase = sm + wave * (4 * ROWP);

  // Stage q[b,dst] and k[b,src] rows (256 floats each) for both jobs.
  // Coalesced: 32 lanes x 16B x 2 per row.
#pragma unroll
  for (int r = 0; r < 4; ++r) {
    const int t = r >> 1;
    const float* grow = (r & 1)
        ? (kk + ((long)bidx[t] * nTok + stok[t]) * ROWF)
        : (q  + ((long)bidx[t] * nTok + dtok[t]) * ROWF);
    float* lrow = wavebase + r * ROWP;
#ifdef HAVE_ASYNC_LDS
    __builtin_amdgcn_global_load_async_to_lds_b128(
        (v4i_vs*)(grow + lane * 4), (v4i_vs*)(lrow + lane * 4), 0, 0);
    __builtin_amdgcn_global_load_async_to_lds_b128(
        (v4i_vs*)(grow + lane * 4 + 128),
        (v4i_vs*)(lrow + lane * 4 + 128), 0, 0);
#else
    *(f32x4*)(lrow + lane * 4)       = *(const f32x4*)(grow + lane * 4);
    *(f32x4*)(lrow + lane * 4 + 128) = *(const f32x4*)(grow + lane * 4 + 128);
#endif
  }
#ifdef HAVE_ASYNC_LDS
  asm volatile("s_wait_asynccnt 0x0" ::: "memory");
#endif

  // Build A (16x32 f16): row m = job*(8)+head, columns = |q-k| over width.
  // ISA 16-bit A layout: lanes 0-15 row m=lane, elems 0..7 -> K0..7,
  // elems 8..15 -> K16..23; lanes 16-31 supply K8..15 / K24..31 of row lane-16.
  const int half = lane >> 4;
  const int r16  = lane & 15;
  const int te   = r16 >> 3;
  const int h    = r16 & 7;
  const float* qp = wavebase + te * (2 * ROWP) + h * WIDTH;
  const float* kp = qp + ROWP;
  const int w0 = half * 8;
  const int w1 = 16 + half * 8;
  v16h a;
#pragma unroll
  for (int jj = 0; jj < 8; ++jj) {
    float dv = qp[w0 + jj] - kp[w0 + jj];
    asm("" : "+v"(dv));   // keep scalar: avoids gfx1250 v2f32-fabs ISel bug
    a[jj] = (_Float16)__builtin_fabsf(dv);
  }
#pragma unroll
  for (int jj = 0; jj < 8; ++jj) {
    float dv = qp[w1 + jj] - kp[w1 + jj];
    asm("" : "+v"(dv));   // keep scalar: avoids gfx1250 v2f32-fabs ISel bug
    a[8 + jj] = (_Float16)__builtin_fabsf(dv);
  }

  v16h ones;
#pragma unroll
  for (int jj = 0; jj < 16; ++jj) ones[jj] = (_Float16)1.0f;

  // D[m,n] = sum_k A[m,k]: all 16 row L1-sums, broadcast across columns.
  v8f c = {};
  v8f d = __builtin_amdgcn_wmma_f32_16x16x32_f16(false, a, false, ones,
                                                 (short)0, c, false, false);

  // D layout: lane 0 holds rows 0..7 (job 0) in d[0..7]; lane 16 rows 8..15.
  if ((lane & 15) == 0) {
    const int t = lane >> 4;
    if (valid[t]) {
      const float scale = -0.17677669529663687f;  // -1/sqrt(32)
      float* Wrow = W + ((long)bidx[t] * E + eidx[t]) * HEADS;
      float* Drow = denom + ((long)bidx[t] * nTok + dtok[t]) * HEADS;
#pragma unroll
      for (int i = 0; i < 8; ++i) {
        float ew = __expf(d[i] * scale);
        Wrow[i] = ew;
        unsafeAtomicAdd(&Drow[i], ew);   // global_atomic_add_f32
      }
    }
  }
}

__global__ void __launch_bounds__(256)
l1sb_scatter_kernel(const float* __restrict__ vf, const float* __restrict__ vb,
                    const int* __restrict__ coo, const float* __restrict__ W,
                    const float* __restrict__ denom, float* __restrict__ vo,
                    int nTok, int E) {
  const int wave = threadIdx.x >> 5;
  const int lane = threadIdx.x & 31;
  const long job = (long)blockIdx.x * WPB + wave;
  const long nJobs = (long)BS * E;
  if (job >= nJobs) return;
  const int b = (int)(job / E);
  const int e = (int)(job - (long)b * E);
  const int dst = coo[3 * e + 0];
  const int src = coo[3 * e + 1];
  const int h  = lane >> 2;           // 8 heads
  const int wq = (lane & 3) * 8;      // 4 lanes x 8 width elems

  const float wgt = W[((long)b * E + e) * HEADS + h] /
                    denom[((long)b * nTok + dst) * HEADS + h];

  const float* vfp = vf + (((long)b * nTok + src) * HEADS + h) * WIDTH + wq;
  const float* vbp = vb + (((long)b * nTok + dst) * HEADS + h) * WIDTH + wq;
  float* voD = vo + (((long)b * nTok + dst) * HEADS + h) * WIDTH + wq;
  float* voS = vo + (((long)b * nTok + src) * HEADS + h) * WIDTH + wq;
#pragma unroll
  for (int j = 0; j < 8; ++j) {
    unsafeAtomicAdd(&voD[j], wgt * vfp[j]);
    unsafeAtomicAdd(&voS[j], wgt * vbp[j]);
  }
}

extern "C" void kernel_launch(void* const* d_in, const int* in_sizes, int n_in,
                              void* d_out, int out_size, void* d_ws, size_t ws_size,
                              hipStream_t stream) {
  const float* vf = (const float*)d_in[0];
  const float* vb = (const float*)d_in[1];
  const float* q  = (const float*)d_in[2];
  const float* k  = (const float*)d_in[3];
  const int*  coo = (const int*)d_in[4];
  // d_in[5] (dst_mxlen) unused: max-free softmax formulation.

  const long E    = (long)in_sizes[4] / 3;
  const long nTok = (long)in_sizes[0] / ((long)BS * HEADS * WIDTH);

  // Workspace: W[bs][E][8] then denom[bs][nTok][8]
  float* W     = (float*)d_ws;
  float* denom = W + (long)BS * E * HEADS;
  float* vo    = (float*)d_out;

  const long nDenom = (long)BS * nTok * HEADS;
  {
    int blocks = (int)(((long)out_size + 255) / 256);
    if (blocks > 2048) blocks = 2048;
    l1sb_zero_kernel<<<blocks, 256, 0, stream>>>(vo, (long)out_size);
  }
  {
    int blocks = (int)((nDenom + 255) / 256);
    if (blocks > 2048) blocks = 2048;
    l1sb_zero_kernel<<<blocks, 256, 0, stream>>>(denom, nDenom);
  }

  const long nJobs = (long)BS * E;
  {
    long grid = (nJobs + (long)(WPB * JPW) - 1) / (WPB * JPW);
    l1sb_ww_kernel<<<(int)grid, 256, 0, stream>>>(q, k, coo, W, denom,
                                                  (int)nTok, (int)E);
  }
  {
    long grid = (nJobs + WPB - 1) / WPB;
    l1sb_scatter_kernel<<<(int)grid, 256, 0, stream>>>(vf, vb, coo, W, denom,
                                                       vo, (int)nTok, (int)E);
  }
}
